// GCN_7327214207513
// MI455X (gfx1250) — matrix operand order repaired and checked by reading, not compiled, not run on hardware
//
#include <hip/hip_runtime.h>

typedef __attribute__((ext_vector_type(16))) _Float16 v16h;
typedef __attribute__((ext_vector_type(8)))  _Float16 v8h;
typedef __attribute__((ext_vector_type(8)))  float    v8f;

#define DIM 128

// ---------------------------------------------------------------------------
// init: agg = b1 (broadcast), out = b2 (broadcast), deg = 1.0 (self loop)
// ---------------------------------------------------------------------------
__global__ __launch_bounds__(256) void k_init(float* __restrict__ agg,
                                              float* __restrict__ out,
                                              float* __restrict__ deg,
                                              const float* __restrict__ b1,
                                              const float* __restrict__ b2,
                                              int n_elem) {
    int idx = blockIdx.x * 256 + threadIdx.x;
    if (idx >= n_elem) return;
    int j = idx & (DIM - 1);
    agg[idx] = b1[j];
    out[idx] = b2[j];
    if (j == 0) deg[idx >> 7] = 1.0f;
}

// deg[col[e]] += 1
__global__ __launch_bounds__(256) void k_deg(const long long* __restrict__ col,
                                             float* __restrict__ deg, int E) {
    int e = blockIdx.x * 256 + threadIdx.x;
    if (e >= E) return;
    unsafeAtomicAdd(&deg[col[e]], 1.0f);
}

// dis = rsqrt(deg)  (in place; deg >= 1 always due to self loops)
__global__ __launch_bounds__(256) void k_dis(float* __restrict__ deg, int n) {
    int i = blockIdx.x * 256 + threadIdx.x;
    if (i >= n) return;
    deg[i] = __frsqrt_rn(deg[i]);
}

// ---------------------------------------------------------------------------
// Convert W [128x128 f32 row-major] into f16 WMMA B-fragment layout:
//   Wf[((ntile*4 + kq)*32 + lane)*16 + h] = W[(kq*32 + 16*(lane>>4) + h)*128
//                                             + ntile*16 + (lane&15)]
// so each lane's 16 B-matrix halves for one 16x16x32 step are contiguous.
// ---------------------------------------------------------------------------
__global__ __launch_bounds__(256) void k_cvt_wfrag(const float* __restrict__ W,
                                                   _Float16* __restrict__ Wf) {
    int idx = blockIdx.x * 256 + threadIdx.x;   // 0 .. 16383
    int h    = idx & 15;
    int lane = (idx >> 4) & 31;
    int kq   = (idx >> 9) & 3;
    int t    = idx >> 11;
    int K = kq * 32 + 16 * (lane >> 4) + h;
    int N = t * 16 + (lane & 15);
    Wf[idx] = (_Float16)W[K * DIM + N];
}

// f32 -> f16 feature conversion (optionally fused ReLU)
template <bool RELU>
__global__ __launch_bounds__(256) void k_cvt_feat(const float* __restrict__ src,
                                                  _Float16* __restrict__ dst,
                                                  int n_elem) {
    int idx = blockIdx.x * 256 + threadIdx.x;
    if (idx >= n_elem) return;
    float v = src[idx];
    if (RELU) v = fmaxf(v, 0.0f);
    dst[idx] = (_Float16)v;
}

// ---------------------------------------------------------------------------
// GEMM: C[M x 128] = A[M x 128] * B[128 x 128], f16 inputs, f32 accumulate.
// One block = 16 rows of A; 8 waves each own one 16-wide N tile.
// K loop: 4 x v_wmma_f32_16x16x32_f16.
// ---------------------------------------------------------------------------
__global__ __launch_bounds__(256) void k_gemm_wmma(const _Float16* __restrict__ Ah,
                                                   const _Float16* __restrict__ Bf,
                                                   float* __restrict__ C, int M) {
    const int lane = threadIdx.x & 31;
    const int wave = threadIdx.x >> 5;        // n-tile 0..7
    const int m0   = blockIdx.x * 16;
    const int lh   = lane >> 4;               // lane half
    const int ml   = lane & 15;
    if (m0 >= M) return;

    const _Float16* arow = Ah + (size_t)(m0 + ml) * DIM;
    v8f c = {};
#pragma unroll
    for (int kq = 0; kq < 4; ++kq) {
        const int k0 = kq * 32;
        // A fragment: K runs [k0+16*lh .. +15]
        v8h alo = *(const v8h*)(arow + k0 + 16 * lh);
        v8h ahi = *(const v8h*)(arow + k0 + 16 * lh + 8);
        v16h a;
#pragma unroll
        for (int i = 0; i < 8; ++i) { a[i] = alo[i]; a[8 + i] = ahi[i]; }
        // B fragment: pre-shuffled, contiguous per lane
        const _Float16* bp = Bf + (((wave * 4 + kq) * 32 + lane) << 4);
        v8h blo = *(const v8h*)(bp);
        v8h bhi = *(const v8h*)(bp + 8);
        v16h b;
#pragma unroll
        for (int i = 0; i < 8; ++i) { b[i] = blo[i]; b[8 + i] = bhi[i]; }
        c = __builtin_amdgcn_wmma_f32_16x16x32_f16(
                /*neg_a=*/false, a, /*neg_b=*/false, b,
                /*c_mod=*/(short)0, c, /*reuse_a=*/false, /*reuse_b=*/false);
    }
    const int n0 = wave * 16;
#pragma unroll
    for (int r = 0; r < 8; ++r)
        C[(size_t)(m0 + r + 8 * lh) * DIM + n0 + ml] = c[r];
}

// ---------------------------------------------------------------------------
// Edge aggregation: one wave per edge. out[col] += dis[row]*dis[col]*h[row]
// Each lane handles 4 consecutive floats (32 lanes * 4 = 128).
// ---------------------------------------------------------------------------
__global__ __launch_bounds__(256) void k_agg_edges(const long long* __restrict__ row,
                                                   const long long* __restrict__ col,
                                                   const float* __restrict__ dis,
                                                   const float* __restrict__ h,
                                                   float* __restrict__ out, int E) {
    const int lane = threadIdx.x & 31;
    const long long e = (long long)blockIdx.x * 8 + (threadIdx.x >> 5);
    if (e >= E) return;
    const long long r = row[e];
    const long long c = col[e];
    const float w = dis[r] * dis[c];
    const float4 v = *(const float4*)(h + r * DIM + lane * 4);
    float* o = out + c * DIM + lane * 4;
    unsafeAtomicAdd(o + 0, w * v.x);
    unsafeAtomicAdd(o + 1, w * v.y);
    unsafeAtomicAdd(o + 2, w * v.z);
    unsafeAtomicAdd(o + 3, w * v.w);
}

// self loops: out[i] += dis[i]^2 * h[i]   (exclusive writer -> plain +=)
__global__ __launch_bounds__(256) void k_agg_self(const float* __restrict__ dis,
                                                  const float* __restrict__ h,
                                                  float* __restrict__ out, int n_elem) {
    int idx = blockIdx.x * 256 + threadIdx.x;
    if (idx >= n_elem) return;
    float d = dis[idx >> 7];
    out[idx] += d * d * h[idx];
}

// ---------------------------------------------------------------------------
extern "C" void kernel_launch(void* const* d_in, const int* in_sizes, int n_in,
                              void* d_out, int out_size, void* d_ws, size_t ws_size,
                              hipStream_t stream) {
    const float*      x   = (const float*)d_in[0];
    const long long*  ei  = (const long long*)d_in[1];   // int64 [2, E]
    const float*      W1  = (const float*)d_in[2];
    const float*      b1  = (const float*)d_in[3];
    const float*      W2  = (const float*)d_in[4];
    const float*      b2  = (const float*)d_in[5];
    float*            out = (float*)d_out;

    const int N = in_sizes[0] / DIM;          // 100000
    const int E = in_sizes[1] / 2;            // 1600000
    const long long NE = (long long)N * DIM;  // 12.8M

    // carve workspace (256B aligned chunks)
    char* p = (char*)d_ws;
    auto carve = [&](size_t bytes) {
        void* r = (void*)p;
        p += (bytes + 255) & ~(size_t)255;
        return r;
    };
    float*     deg  = (float*)carve((size_t)N * 4);            // deg -> dis in place
    _Float16*  fh16 = (_Float16*)carve((size_t)NE * 2);        // f16 features (both layers)
    float*     h    = (float*)carve((size_t)NE * 4);           // GEMM output (both layers)
    float*     agg  = (float*)carve((size_t)NE * 4);           // layer1 aggregate
    _Float16*  Wf1  = (_Float16*)carve((size_t)DIM * DIM * 2); // W1 fragments
    _Float16*  Wf2  = (_Float16*)carve((size_t)DIM * DIM * 2); // W2 fragments

    const int gElem  = (int)((NE + 255) / 256);   // 50000
    const int gEdge  = (E + 255) / 256;
    const int gNode  = (N + 255) / 256;
    const int gGemm  = (N + 15) / 16;             // 6250
    const int gWaveE = (E + 7) / 8;               // one wave per edge

    // degrees / normalization / bias-folded accumulator init
    k_init<<<gElem, 256, 0, stream>>>(agg, out, deg, b1, b2, (int)NE);
    k_deg<<<gEdge, 256, 0, stream>>>(ei + E, deg, E);
    k_dis<<<gNode, 256, 0, stream>>>(deg, N);

    // weight fragmentization + x -> f16
    k_cvt_wfrag<<<64, 256, 0, stream>>>(W1, Wf1);
    k_cvt_wfrag<<<64, 256, 0, stream>>>(W2, Wf2);
    k_cvt_feat<false><<<gElem, 256, 0, stream>>>(x, fh16, (int)NE);

    // layer 1: h = x @ W1 ; agg = b1 + scatter_add(norm * h[row])
    k_gemm_wmma<<<gGemm, 256, 0, stream>>>(fh16, Wf1, h, N);
    k_agg_self<<<gElem, 256, 0, stream>>>(deg, h, agg, (int)NE);
    k_agg_edges<<<gWaveE, 256, 0, stream>>>(ei, ei + E, deg, h, agg, E);

    // relu + f16 convert, layer 2
    k_cvt_feat<true><<<gElem, 256, 0, stream>>>(agg, fh16, (int)NE);
    k_gemm_wmma<<<gGemm, 256, 0, stream>>>(fh16, Wf2, h, N);
    k_agg_self<<<gElem, 256, 0, stream>>>(deg, h, out, (int)NE);
    k_agg_edges<<<gWaveE, 256, 0, stream>>>(ei, ei + E, deg, h, out, E);
}
